// MsaNet_50371376447847
// MI455X (gfx1250) — compile-verified, matrix operand order checked by reading, not run on hardware
//
#include <hip/hip_runtime.h>
#include <hip/hip_bf16.h>

// ---------------------------------------------------------------------------
// MsaNet ROI head for MI455X (gfx1250, wave32, WMMA).
// One workgroup per proposal; all intermediates live in LDS (~185 KB of the
// 320 KB WGP LDS); both 1x1 convs run on v_wmma_f32_16x16x32_f16.
//  - conv1x1 commutes with trilinear upsampling => GEMM1 on the 7^3 grid,
//    upsampling done on 48 channels out of LDS.
//  - crop & weights stored in LDS pre-swizzled into WMMA fragment order:
//    one v16h (2x ds_load_b128) per operand fragment, no scalar gathers.
//  - GEMM2 B fragments built directly in registers (interp + out1 loads),
//    no LDS staging round-trip.
//  - GEMM2 computed twice (stats pass + output pass): no global scratch.
// ---------------------------------------------------------------------------

typedef __attribute__((ext_vector_type(16))) _Float16 v16h;
typedef __attribute__((ext_vector_type(8)))  float    v8f;

#define WMMA_F16(a, b, c) \
  __builtin_amdgcn_wmma_f32_16x16x32_f16(false, (a), false, (b), (short)0, (c), false, false)

constexpr int NPROP   = 64;
constexpr int C2c     = 128;   // comb2 channels
constexpr int C1c     = 48;    // out1 channels
constexpr int S2      = 7;     // comb2 crop side
constexpr int S1      = 14;    // out1 crop side / upsampled side
constexpr int NS      = S2 * S2 * S2;   // 343
constexpr int NU      = S1 * S1 * S1;   // 2744
constexpr int NSP     = 352;   // NS padded to /16
constexpr int NUP     = 2752;  // NU padded to /16
constexpr int NT1     = NSP / 16;       // 22 N-tiles, GEMM1
constexpr int NT2     = NUP / 16;       // 172 N-tiles, GEMM2
constexpr int THREADS = 256;
constexpr int NWAVE   = 8;
constexpr float EPSV  = 1e-5f;

// ---- LDS layout (dynamic shared; ~184.8 KB total) -------------------------
constexpr int OFF_CROP = 0;
constexpr int SZ_CROP  = C2c * NSP * 2;            // 90112 B f16 crop, frag-ordered
                                                   // [kt=4][nt=22][lane=32][e=16]
                                                   // (aliased by pool f32[64*343] after GEMM1)
constexpr int OFF_YS   = OFF_CROP + SZ_CROP;
constexpr int SZ_YS    = C1c * NSP * 4;            // 67584 B f32 y_small [48][NSP]
constexpr int OFF_WUP  = OFF_YS + SZ_YS;
constexpr int SZ_WUP   = C1c * C2c * 2;            // 12288 B f16 W_up2 frag-ordered [3][4][32][16]
constexpr int OFF_WBK  = OFF_WUP + SZ_WUP;
constexpr int SZ_WBK   = 64 * 96 * 2;              // 12288 B f16 W_back2 frag-ordered [4][3][32][16]
constexpr int OFF_F    = OFF_WBK + SZ_WBK;
constexpr int SZ_F     = 640 * 4;                  // small float/int arrays
constexpr int SMEM_BYTES = OFF_F + SZ_F;           // 184832 B

// fragment-order scatter index: element (row r in 0..15-block, k within 32)
// forward map (ISA 7.12.2): k = e + 8*(e>=8) + 8*(lane>=16), row = lane&15
__device__ __forceinline__ int frag_scatter(int kk /*0..31*/, int row /*0..15*/,
                                            int* lane_out) {
  int hi  = (kk >> 3) & 1;        // k bit 3 selects the half-wave
  int kk2 = kk - 8 * hi;          // in {0..7} u {16..23}
  int e   = (kk2 >= 16) ? kk2 - 8 : kk2;
  *lane_out = row + 16 * hi;
  return e;
}

// trilinear sample of one 7^3 channel held in LDS ([z*49+y*7+x], row ld NSP)
__device__ __forceinline__ float interp1(const float* ysc, int uz, int uy, int ux,
                                         const int* lo, const float* fr) {
  int lz = lo[uz], ly = lo[uy], lx = lo[ux];
  float fz = fr[uz], fy = fr[uy], fx = fr[ux];
  const float* b = ysc + lz * 49 + ly * 7 + lx;
  float c00 = b[0]  * (1.f - fx) + b[1]  * fx;
  float c01 = b[7]  * (1.f - fx) + b[8]  * fx;
  float c10 = b[49] * (1.f - fx) + b[50] * fx;
  float c11 = b[56] * (1.f - fx) + b[57] * fx;
  float c0 = c00 * (1.f - fy) + c01 * fy;
  float c1 = c10 * (1.f - fy) + c11 * fy;
  return c0 * (1.f - fz) + c1 * fz;
}

__global__ void __launch_bounds__(THREADS, 1)
msanet_roi_kernel(const float* __restrict__ out1, const float* __restrict__ comb2,
                  const int* __restrict__ proposals,
                  const float* __restrict__ w_up2, const float* __restrict__ b_up2,
                  const float* __restrict__ g_up2, const float* __restrict__ be_up2,
                  const float* __restrict__ w_bk2, const float* __restrict__ b_bk2,
                  const float* __restrict__ g_bk2, const float* __restrict__ be_bk2,
                  float* __restrict__ out) {
  extern __shared__ __align__(32) char smem[];
  _Float16* crop = (_Float16*)(smem + OFF_CROP);        // frag-ordered f16
  float*    pool = (float*)(smem + OFF_CROP);           // alias, [64][343] f32
  float*    ys   = (float*)(smem + OFF_YS);             // [48][NSP] f32
  _Float16* wup  = (_Float16*)(smem + OFF_WUP);         // frag-ordered
  _Float16* wbk  = (_Float16*)(smem + OFF_WBK);         // frag-ordered
  float*    fb   = (float*)(smem + OFF_F);
  float* bias1 = fb +   0;  float* sum1 = fb +  48;  float* sq1 = fb +  96;
  float* sc1   = fb + 144;  float* sh1  = fb + 192;
  float* bias2 = fb + 240;  float* sum2 = fb + 304;  float* sq2 = fb + 368;
  float* sc2   = fb + 432;  float* sh2  = fb + 496;
  float* upfr  = fb + 560;                              // [14]
  int*   uplo  = (int*)(fb + 576);                      // [14]

  const int tid  = threadIdx.x;
  const int lane = tid & 31;
  const int wave = tid >> 5;
  const int prop = blockIdx.x;
  const int hi8  = (lane >= 16) ? 8 : 0;

  const int bidx = proposals[prop * 7 + 0];
  const int z0   = proposals[prop * 7 + 1];
  const int y0   = proposals[prop * 7 + 2];
  const int x0   = proposals[prop * 7 + 3];
  const int cz = z0 >> 2, cy = y0 >> 2, cx = x0 >> 2;      // comb2 crop origin
  const int oz = z0 >> 1, oy = y0 >> 1, ox = x0 >> 1;      // out1 crop origin

  // ---- stage 0: tables, weights (frag-ordered), crop (frag-ordered) -------
  if (tid < S1) {
    float pos = (float)tid * (float)(S2 - 1) / (float)(S1 - 1);
    int lo = (int)floorf(pos);
    if (lo > S2 - 2) lo = S2 - 2;
    uplo[tid] = lo;
    upfr[tid] = pos - (float)lo;
  }
  for (int i = tid; i < C1c * C2c; i += THREADS) {       // W_up2 [48][128]
    int m = i / C2c, k = i - m * C2c;
    int ln, e = frag_scatter(k & 31, m & 15, &ln);
    wup[((((m >> 4) * 4) + (k >> 5)) * 32 + ln) * 16 + e] = (_Float16)w_up2[i];
  }
  for (int i = tid; i < 64 * 96; i += THREADS) {         // W_back2 [64][96]
    int m = i / 96, k = i - m * 96;
    int ln, e = frag_scatter(k & 31, m & 15, &ln);
    wbk[((((m >> 4) * 3) + (k >> 5)) * 32 + ln) * 16 + e] = (_Float16)w_bk2[i];
  }
  if (tid < C1c) { bias1[tid] = b_up2[tid]; sum1[tid] = 0.f; sq1[tid] = 0.f; }
  if (tid < 64)  { bias2[tid] = b_bk2[tid]; sum2[tid] = 0.f; sq2[tid] = 0.f; }

  for (int i = tid; i < C2c * NSP; i += THREADS) {       // comb2 crop [128][NSP]
    int k = i / NSP, n = i - k * NSP;
    float v = 0.f;
    if (n < NS) {
      int dz = n / 49, r = n - dz * 49, dy = r / 7, dx = r - dy * 7;
      v = comb2[(((size_t)(bidx * C2c + k) * 32 + (cz + dz)) * 32 + (cy + dy)) * 32 + (cx + dx)];
    }
    int ln, e = frag_scatter(k & 31, n & 15, &ln);
    crop[((((k >> 5) * NT1) + (n >> 4)) * 32 + ln) * 16 + e] = (_Float16)v;
  }
  __syncthreads();

  // ---- stage 1: GEMM1  y_small[48,343] = W_up2 @ crop + b -----------------
  {
    const v16h* wupf  = (const v16h*)wup;
    const v16h* cropf = (const v16h*)crop;
    for (int t = wave; t < 3 * NT1; t += NWAVE) {
      int mt = t / NT1, nt = t - mt * NT1;
      v8f acc = {};
#pragma unroll
      for (int kt = 0; kt < 4; ++kt)
        acc = WMMA_F16(wupf[(mt * 4 + kt) * 32 + lane],
                       cropf[(kt * NT1 + nt) * 32 + lane], acc);
      int nn = nt * 16 + (lane & 15);
#pragma unroll
      for (int r = 0; r < 8; ++r) {
        int m = mt * 16 + r + hi8;
        ys[m * NSP + nn] = acc[r] + bias1[m];
      }
    }
  }
  __syncthreads();

  // ---- stage 2: InstanceNorm stats of upsampled y (14^3), wave-reduced ----
  for (int c = 0; c < C1c; ++c) {
    const float* yc = ys + c * NSP;
    float s = 0.f, q = 0.f;
    for (int p = tid; p < NU; p += THREADS) {
      int uz = p / 196, r = p - uz * 196, uy = r / 14, ux = r - uy * 14;
      float v = interp1(yc, uz, uy, ux, uplo, upfr);
      s += v;
      q += v * v;
    }
#pragma unroll
    for (int off = 16; off > 0; off >>= 1) {
      s += __shfl_down(s, off, 32);
      q += __shfl_down(q, off, 32);
    }
    if (lane == 0) { atomicAdd(&sum1[c], s); atomicAdd(&sq1[c], q); }
  }
  __syncthreads();
  if (tid < C1c) {
    float mean = sum1[tid] * (1.f / (float)NU);
    float var  = sq1[tid] * (1.f / (float)NU) - mean * mean;
    float rstd = rsqrtf(var + EPSV);
    float s = g_up2[tid] * rstd;
    sc1[tid] = s;
    sh1[tid] = be_up2[tid] - s * mean;
  }
  __syncthreads();

  // ---- stage 3/4: GEMM2 twice (stats pass, then output pass) --------------
  v16h a2[4][3];
  {
    const v16h* wbkf = (const v16h*)wbk;
#pragma unroll
    for (int mt = 0; mt < 4; ++mt)
#pragma unroll
      for (int kt = 0; kt < 3; ++kt)
        a2[mt][kt] = wbkf[(mt * 3 + kt) * 32 + lane];
  }
  const float* o1base = out1 + (size_t)bidx * C1c * 64 * 64 * 64;

  // B fragments built directly in registers: lane owns spatial n = lane&15,
  // its 48 (k,n) elements = 24 interp channels + 24 out1 channels.
  auto gemm2_tile = [&](int nt, v8f acc[4]) {
    int p = nt * 16 + (lane & 15);
    bool valid = p < NU;
    float w000 = 0.f, w001 = 0.f, w010 = 0.f, w011 = 0.f;
    float w100 = 0.f, w101 = 0.f, w110 = 0.f, w111 = 0.f;
    int b0 = 0, oofs = 0;
    if (valid) {
      int uz = p / 196, r = p - uz * 196, uy = r / 14, ux = r - uy * 14;
      int lz = uplo[uz], ly = uplo[uy], lx = uplo[ux];
      float fz = upfr[uz], fy = upfr[uy], fx = upfr[ux];
      b0 = lz * 49 + ly * 7 + lx;
      float gz = 1.f - fz, gy = 1.f - fy, gx = 1.f - fx;
      w000 = gz * gy * gx; w001 = gz * gy * fx;
      w010 = gz * fy * gx; w011 = gz * fy * fx;
      w100 = fz * gy * gx; w101 = fz * gy * fx;
      w110 = fz * fy * gx; w111 = fz * fy * fx;
      oofs = ((oz + uz) * 64 + (oy + uy)) * 64 + (ox + ux);
    }
#pragma unroll
    for (int kt = 0; kt < 3; ++kt) {
      v16h bf;
#pragma unroll
      for (int e = 0; e < 16; ++e) {
        int k = kt * 32 + ((e < 8) ? e : e + 8) + hi8;
        float v = 0.f;
        if (valid) {
          if (k < C1c) {
            const float* yc = ys + k * NSP + b0;
            float t = w000 * yc[0]  + w001 * yc[1]  + w010 * yc[7]  + w011 * yc[8] +
                      w100 * yc[49] + w101 * yc[50] + w110 * yc[56] + w111 * yc[57];
            t = t * sc1[k] + sh1[k];
            v = fmaxf(t, 0.f);
          } else {
            v = o1base[(size_t)(k - C1c) * (64 * 64 * 64) + oofs];
          }
        }
        bf[e] = (_Float16)v;
      }
#pragma unroll
      for (int mt = 0; mt < 4; ++mt) acc[mt] = WMMA_F16(a2[mt][kt], bf, acc[mt]);
    }
  };

  // pass 1: per-channel stats of z = W_back2 @ B + b over valid positions
  float lsum[4][8], lsq[4][8];
#pragma unroll
  for (int mt = 0; mt < 4; ++mt)
#pragma unroll
    for (int r = 0; r < 8; ++r) { lsum[mt][r] = 0.f; lsq[mt][r] = 0.f; }

  for (int nt = wave; nt < NT2; nt += NWAVE) {
    v8f acc[4] = {};
    gemm2_tile(nt, acc);
    int p = nt * 16 + (lane & 15);
    if (p < NU) {
#pragma unroll
      for (int mt = 0; mt < 4; ++mt)
#pragma unroll
        for (int r = 0; r < 8; ++r) {
          float z = acc[mt][r] + bias2[mt * 16 + r + hi8];
          lsum[mt][r] += z;
          lsq[mt][r]  += z * z;
        }
    }
  }
#pragma unroll
  for (int mt = 0; mt < 4; ++mt)
#pragma unroll
    for (int r = 0; r < 8; ++r) {
      int m = mt * 16 + r + hi8;
      atomicAdd(&sum2[m], lsum[mt][r]);
      atomicAdd(&sq2[m],  lsq[mt][r]);
    }
  __syncthreads();

  if (tid < 64) {
    float mean = sum2[tid] * (1.f / (float)NU);
    float var  = sq2[tid] * (1.f / (float)NU) - mean * mean;
    float rstd = rsqrtf(var + EPSV);
    float s = g_bk2[tid] * rstd;
    sc2[tid] = s;
    sh2[tid] = be_bk2[tid] - s * mean;
  }
  // pool buffer aliases the (now dead) crop region; ReLU output >= 0 so init 0
  for (int i = tid; i < 64 * NS; i += THREADS) pool[i] = 0.f;
  __syncthreads();

  // pass 2: recompute, normalize + relu, 2x2x2 maxpool via ds_max_u32
  for (int nt = wave; nt < NT2; nt += NWAVE) {
    v8f acc[4] = {};
    gemm2_tile(nt, acc);
    int p = nt * 16 + (lane & 15);
    if (p < NU) {
      int uz = p / 196, r2 = p - uz * 196, uy = r2 / 14, ux = r2 - uy * 14;
      int pidx = (uz >> 1) * 49 + (uy >> 1) * 7 + (ux >> 1);
#pragma unroll
      for (int mt = 0; mt < 4; ++mt)
#pragma unroll
        for (int r = 0; r < 8; ++r) {
          int m = mt * 16 + r + hi8;
          float z = acc[mt][r] + bias2[m];
          float t = z * sc2[m] + sh2[m];
          t = fmaxf(t, 0.f);
          // non-negative floats order like their unsigned bit patterns
          atomicMax((unsigned int*)&pool[m * NS + pidx], __float_as_uint(t));
        }
    }
  }
  __syncthreads();

  // ---- stage 5: write [64, 7,7,7] result ----------------------------------
  float* op = out + (size_t)prop * 64 * NS;
  for (int i = tid; i < 64 * NS; i += THREADS) op[i] = pool[i];
}

extern "C" void kernel_launch(void* const* d_in, const int* in_sizes, int n_in,
                              void* d_out, int out_size, void* d_ws, size_t ws_size,
                              hipStream_t stream) {
  (void)in_sizes; (void)n_in; (void)out_size; (void)d_ws; (void)ws_size;
  const float* out1   = (const float*)d_in[0];
  const float* comb2  = (const float*)d_in[1];
  const int*   props  = (const int*)d_in[2];
  const float* w_up2  = (const float*)d_in[3];
  const float* b_up2  = (const float*)d_in[4];
  const float* g_up2  = (const float*)d_in[5];
  const float* be_up2 = (const float*)d_in[6];
  const float* w_bk2  = (const float*)d_in[7];
  const float* b_bk2  = (const float*)d_in[8];
  const float* g_bk2  = (const float*)d_in[9];
  const float* be_bk2 = (const float*)d_in[10];
  float* out = (float*)d_out;

  // allow > default dynamic LDS (CDNA5 WGP: 320 KB); deterministic, not a
  // stream operation (graph-capture safe).
  hipFuncSetAttribute((const void*)msanet_roi_kernel,
                      hipFuncAttributeMaxDynamicSharedMemorySize, SMEM_BYTES);

  msanet_roi_kernel<<<dim3(NPROP), dim3(THREADS), SMEM_BYTES, stream>>>(
      out1, comb2, props, w_up2, b_up2, g_up2, be_up2,
      w_bk2, b_bk2, g_bk2, be_bk2, out);
}
